// Transformer_83992380441442
// MI455X (gfx1250) — matrix operand order repaired and checked by reading, not compiled, not run on hardware
//
#include <hip/hip_runtime.h>

typedef __attribute__((ext_vector_type(16))) _Float16 v16h;
typedef __attribute__((ext_vector_type(8)))  float    v8f;

// ---------------------------------------------------------------------------
// Prep 1: im2col index table  idxTab[k] = (c*Hin + rr)*Win + ss  (0 for k>=K)
// ---------------------------------------------------------------------------
__global__ __launch_bounds__(128) void build_idx_kernel(
    int* __restrict__ idx, int K, int Kpad, int KHW, int KW, int Hin, int Win)
{
  int k = blockIdx.x * 128 + threadIdx.x;
  if (k >= Kpad) return;
  int v = 0;
  if (k < K) {
    int c  = k / KHW;
    int r2 = k - c * KHW;
    int rr = r2 / KW;
    int ss = r2 - rr * KW;
    v = (c * Hin + rr) * Win + ss;
  }
  idx[k] = v;
}

// ---------------------------------------------------------------------------
// Prep 2: swizzle OIHW weights (N x K) into zero-padded f16 B-fragment order:
//   dst[((tile*nChunks + chunk)*32 + lane)*16 + e] = w[tile*16+(lane&15)]
//                                                    [chunk*32 + e + 16*(lane>>4)]
// so each wave-lane reads its 16 B f16 values as one contiguous 32B chunk.
// Zero padding for k>=K / n>=Cout nullifies any garbage in the A tail.
// ---------------------------------------------------------------------------
__global__ __launch_bounds__(256) void build_wswz_kernel(
    _Float16* __restrict__ dst, const float* __restrict__ w,
    int Cout, int K, int nChunks, int nTilesPad)
{
  int total = nTilesPad * nChunks * 512;
  int g = blockIdx.x * 256 + threadIdx.x;
  if (g >= total) return;
  int e     = g & 15;
  int L     = (g >> 4) & 31;
  int rest  = g >> 9;
  int chunk = rest % nChunks;
  int tile  = rest / nChunks;
  int n = tile * 16 + (L & 15);
  int k = chunk * 32 + e + 16 * (L >> 4);
  float v = (n < Cout && k < K) ? w[(size_t)n * K + k] : 0.f;
  dst[g] = (_Float16)v;
}

// ---------------------------------------------------------------------------
// Implicit-GEMM conv: block = 4 waves sharing one 32-pixel M tile (two 16-row
// A fragments), covering 4 adjacent 16-channel N tiles. A chunks (2 x 16x32
// f16) are gathered once per block into LDS (fragment order, double buffered,
// idxTab quad shared by both sub-tiles); B comes pre-swizzled + prefetched
// one chunk ahead (global_prefetch_b8). f16 WMMA, f32 accumulate. No
// divergence around the WMMAs -> EXEC all-ones as required.
// ---------------------------------------------------------------------------
__global__ __launch_bounds__(128) void conv_wmma_tiled(
    const float* __restrict__ in, const int* __restrict__ idxTab,
    const _Float16* __restrict__ wSwz, const float* __restrict__ bias,
    float* __restrict__ out,
    int M, int HWout, int Wout, int Cout, int nChunks,
    int inBatchStride, int Win, int stride, int doRelu)
{
  const int tid  = threadIdx.x;
  const int wave = tid >> 5;
  const int lane = tid & 31;
  const int half = lane >> 4;
  const int lm   = lane & 15;
  const int tileM = blockIdx.x * 32;

  // ---- loader role: thread fills both sub-tiles at [L*16 + 4u .. +3]
  const int L     = tid >> 2;
  const int u     = tid & 3;
  const int halfL = L >> 4;

  auto rowBase = [&](int m) -> const float* {
    if (m > M - 1) m = M - 1;            // clamp: garbage rows are never stored
    int b  = m / HWout;
    int rp = m - b * HWout;
    int oh = rp / Wout;
    int ow = rp - oh * Wout;
    return in + (size_t)b * inBatchStride
              + (size_t)(oh * stride) * Win + (size_t)(ow * stride);
  };
  const float* aBase0 = rowBase(tileM + (L & 15));
  const float* aBase1 = rowBase(tileM + 16 + (L & 15));

  // A fragment element e maps to k-offset: e + 16*half.
  // This thread's 4 elements e = 4u..4u+3 are 4 *consecutive* k's:
  const int kOffBase = 4 * u + 16 * halfL;

  __shared__ __align__(32) _Float16 sA[2][2][512];   // [buf][subtile][frag]

  const _Float16* bTile = wSwz + (size_t)(blockIdx.y * 4 + wave) * nChunks * 512;

  v8f acc0 = {}, acc1 = {};
  for (int chunk = 0; chunk < nChunks; ++chunk) {
    const int* ip = idxTab + chunk * 32 + kOffBase;
    int i0 = ip[0], i1 = ip[1], i2 = ip[2], i3 = ip[3];
    _Float16* d0 = &sA[chunk & 1][0][L * 16 + 4 * u];
    _Float16* d1 = &sA[chunk & 1][1][L * 16 + 4 * u];
    d0[0] = (_Float16)aBase0[i0];
    d0[1] = (_Float16)aBase0[i1];
    d0[2] = (_Float16)aBase0[i2];
    d0[3] = (_Float16)aBase0[i3];
    d1[0] = (_Float16)aBase1[i0];
    d1[1] = (_Float16)aBase1[i1];
    d1[2] = (_Float16)aBase1[i2];
    d1[3] = (_Float16)aBase1[i3];
    // Prefetch next chunk's B fragment (clamped -> always a valid address).
    int pc = (chunk + 1 < nChunks) ? (chunk + 1) : chunk;
    __builtin_prefetch(bTile + (size_t)pc * 512 + lane * 16, 0, 3);
    __syncthreads();   // double buffer: one barrier per chunk is sufficient
    v16h bF = *(const v16h*)(bTile + (size_t)chunk * 512 + lane * 16);
    v16h a0 = *(const v16h*)&sA[chunk & 1][0][lane * 16];
    v16h a1 = *(const v16h*)&sA[chunk & 1][1][lane * 16];
    acc0 = __builtin_amdgcn_wmma_f32_16x16x32_f16(false, a0, false, bF,
                                                  (short)0, acc0, false, false);
    acc1 = __builtin_amdgcn_wmma_f32_16x16x32_f16(false, a1, false, bF,
                                                  (short)0, acc1, false, false);
  }

  // D layout: VGPR r -> (row mBase + r + 8*half, col lm)
  int nOut = (blockIdx.y * 4 + wave) * 16 + lm;
  if (nOut < Cout) {
    float bb = bias[nOut];
    auto storeTile = [&](const v8f& acc, int mBase) {
#pragma unroll
      for (int r = 0; r < 8; ++r) {
        int mOut = mBase + r + half * 8;
        if (mOut < M) {
          int b_ = mOut / HWout;
          int pp = mOut - b_ * HWout;
          float v = acc[r] + bb;
          if (doRelu) v = v > 0.f ? v : 0.f;
          out[((size_t)b_ * Cout + nOut) * HWout + pp] = v;
        }
      }
    };
    storeTile(acc0, tileM);
    storeTile(acc1, tileM + 16);
  }
}

// ---------------------------------------------------------------------------
// In-place instance norm (population var, eps=1e-5) + ReLU; one block per (n,c).
// ---------------------------------------------------------------------------
__global__ __launch_bounds__(256) void inorm_relu_kernel(float* __restrict__ data, int HW)
{
  float* p = data + (size_t)blockIdx.x * HW;
  __shared__ float s1[256], s2[256];
  float sum = 0.f, sq = 0.f;
  for (int i = threadIdx.x; i < HW; i += 256) {
    float v = p[i];
    sum += v; sq += v * v;
  }
  s1[threadIdx.x] = sum; s2[threadIdx.x] = sq;
  __syncthreads();
  for (int s = 128; s > 0; s >>= 1) {
    if (threadIdx.x < s) {
      s1[threadIdx.x] += s1[threadIdx.x + s];
      s2[threadIdx.x] += s2[threadIdx.x + s];
    }
    __syncthreads();
  }
  float mean = s1[0] / (float)HW;
  float var  = s2[0] / (float)HW - mean * mean;
  float inv  = rsqrtf(var + 1e-5f);
  for (int i = threadIdx.x; i < HW; i += 256) {
    float v = (p[i] - mean) * inv;
    p[i] = v > 0.f ? v : 0.f;
  }
}

// ---------------------------------------------------------------------------
// Head: (2,3,81) -> fc1(81->72,relu) -> fc2(72->65) -> folded 5x-concat @ w_l.
// ---------------------------------------------------------------------------
__global__ __launch_bounds__(512) void head_kernel(
    const float* __restrict__ fin,
    const float* __restrict__ w_c1, const float* __restrict__ b_c1,
    const float* __restrict__ w_c2, const float* __restrict__ b_c2,
    const float* __restrict__ w_l,  const float* __restrict__ b_l,
    float* __restrict__ h_theta)
{
  __shared__ float sFin[486];
  __shared__ float sF1[432];
  __shared__ float sB[390];
  __shared__ float sW[4225];
  int tid = threadIdx.x;
  for (int i = tid; i < 486; i += 512) sFin[i] = fin[i];
  __syncthreads();
  if (tid < 432) {
    int row = tid / 72, o = tid % 72;
    float s = b_c1[o];
    for (int k = 0; k < 81; ++k) s += sFin[row * 81 + k] * w_c1[o * 81 + k];
    sF1[tid] = s > 0.f ? s : 0.f;
  }
  __syncthreads();
  if (tid < 390) {
    int row = tid / 65, o = tid % 65;
    float s = b_c2[o];
    for (int k = 0; k < 72; ++k) s += sF1[row * 72 + k] * w_c2[o * 72 + k];
    sB[tid] = s;
  }
  __syncthreads();
  for (int i = tid; i < 4225; i += 512) {
    int o = i / 65, j = i - o * 65;
    float s = 0.f;
    for (int t = 0; t < 5; ++t) s += w_l[o * 325 + t * 65 + j];
    sW[i] = s;
  }
  __syncthreads();
  if (tid < 390) {
    int row = tid / 65, o = tid % 65;     // row = b*3 + c
    float s = b_l[o];
    for (int j = 0; j < 65; ++j) s += sB[row * 65 + j] * sW[o * 65 + j];
    h_theta[tid] = s;
  }
}

// ---------------------------------------------------------------------------
// y1[b,f,:] = sum_a {R,G,B,R2,G2,B2,RG,GB,BR,1}_a * h_theta[b][f][a], a<10
// ---------------------------------------------------------------------------
__global__ __launch_bounds__(256) void y1_kernel(
    const float* __restrict__ y, const float* __restrict__ h_theta,
    float* __restrict__ y1)
{
  __shared__ float sH2[60];
  if (threadIdx.x < 60) {
    int b_ = threadIdx.x / 30, r = threadIdx.x % 30, f = r / 10, a = r % 10;
    sH2[threadIdx.x] = h_theta[(b_ * 3 + f) * 65 + a];
  }
  __syncthreads();
  int idx = blockIdx.x * 256 + threadIdx.x;
  if (idx >= 2 * 512 * 512) return;
  int b_ = idx >> 18;
  int p  = idx & 262143;
  float R = y[((size_t)(b_ * 3 + 0)) * 262144 + p];
  float G = y[((size_t)(b_ * 3 + 1)) * 262144 + p];
  float B = y[((size_t)(b_ * 3 + 2)) * 262144 + p];
  float feat[10] = {R, G, B, R * R, G * G, B * B, R * G, G * B, B * R, 1.f};
#pragma unroll
  for (int f = 0; f < 3; ++f) {
    const float* cf = &sH2[b_ * 30 + f * 10];
    float s = 0.f;
#pragma unroll
    for (int a = 0; a < 10; ++a) s += feat[a] * cf[a];
    y1[((size_t)(b_ * 3 + f)) * 262144 + p] = s;
  }
}

// ---------------------------------------------------------------------------
// out[b,f,i,j] = sum_a (sum_c feat_a(y1[b,c],i,j)) * h_theta[b][f][10+a] + y
// _shift(x,di,dj)[i,j] = x[i-di, j-dj] with zero padding.
// ---------------------------------------------------------------------------
__global__ __launch_bounds__(256) void final_kernel(
    const float* __restrict__ y1, const float* __restrict__ y,
    const float* __restrict__ h_theta, float* __restrict__ out)
{
  __shared__ float sH1[330];
  for (int i = threadIdx.x; i < 330; i += 256) {
    int b_ = i / 165, r = i - b_ * 165, f = r / 55, a = r - f * 55;
    sH1[i] = h_theta[(b_ * 3 + f) * 65 + 10 + a];
  }
  __syncthreads();
  int idx = blockIdx.x * 256 + threadIdx.x;
  if (idx >= 2 * 512 * 512) return;
  int b_ = idx >> 18;
  int p  = idx & 262143;
  int i  = p >> 9, j = p & 511;

  float fs[55];
#pragma unroll
  for (int a = 0; a < 55; ++a) fs[a] = 0.f;

#pragma unroll
  for (int c = 0; c < 3; ++c) {
    const float* ch = y1 + ((size_t)(b_ * 3 + c)) * 262144;
    float xv  = ch[p];
    float Rv  = (j > 0)              ? ch[p - 1]   : 0.f;
    float Lv  = (j < 511)            ? ch[p + 1]   : 0.f;
    float Uv  = (i > 0)              ? ch[p - 512] : 0.f;
    float Dv  = (i < 511)            ? ch[p + 512] : 0.f;
    float RUv = (i > 0   && j > 0)   ? ch[p - 513] : 0.f;
    float RDv = (i < 511 && j > 0)   ? ch[p + 511] : 0.f;
    float LUv = (i > 0   && j < 511) ? ch[p - 511] : 0.f;
    float LDv = (i < 511 && j < 511) ? ch[p + 513] : 0.f;
    int a = 0;
    fs[a++] += 1.f;
    fs[a++] += xv;  fs[a++] += Rv;  fs[a++] += Lv;  fs[a++] += Uv;  fs[a++] += Dv;
    fs[a++] += RDv; fs[a++] += RUv; fs[a++] += LUv; fs[a++] += LDv;
    fs[a++] += xv * xv;   fs[a++] += Rv * Rv;   fs[a++] += Lv * Lv;
    fs[a++] += Uv * Uv;   fs[a++] += Dv * Dv;   fs[a++] += RUv * RUv;
    fs[a++] += RDv * RDv; fs[a++] += LUv * LUv; fs[a++] += LDv * LDv;
    fs[a++] += xv * Rv;  fs[a++] += xv * Lv;  fs[a++] += xv * Uv;  fs[a++] += xv * Dv;
    fs[a++] += xv * RUv; fs[a++] += xv * RDv; fs[a++] += xv * LUv; fs[a++] += xv * LDv;
    fs[a++] += Rv * Uv;  fs[a++] += Rv * Dv;  fs[a++] += Rv * Lv;  fs[a++] += Rv * RUv;
    fs[a++] += Rv * RDv; fs[a++] += Rv * LUv; fs[a++] += Rv * LDv;
    fs[a++] += Uv * Dv;  fs[a++] += Uv * Lv;  fs[a++] += Uv * RUv; fs[a++] += Uv * RDv;
    fs[a++] += Uv * LUv; fs[a++] += Uv * LDv;
    fs[a++] += Dv * Lv;  fs[a++] += Dv * RUv; fs[a++] += Dv * RDv; fs[a++] += Dv * LUv;
    fs[a++] += Dv * LDv;
    fs[a++] += Lv * RUv; fs[a++] += Lv * RDv; fs[a++] += Lv * LUv; fs[a++] += Lv * LDv;
    fs[a++] += RUv * RDv; fs[a++] += RUv * LUv; fs[a++] += RUv * LDv;
    fs[a++] += RDv * LUv; fs[a++] += RDv * LDv;
    fs[a++] += LUv * LDv;
  }

#pragma unroll
  for (int f = 0; f < 3; ++f) {
    const float* cf = &sH1[b_ * 165 + f * 55];
    float s = 0.f;
#pragma unroll
    for (int a = 0; a < 55; ++a) s += fs[a] * cf[a];
    size_t o = ((size_t)(b_ * 3 + f)) * 262144 + p;
    out[o] = s + y[o];
  }
}

// ---------------------------------------------------------------------------
extern "C" void kernel_launch(void* const* d_in, const int* in_sizes, int n_in,
                              void* d_out, int out_size, void* d_ws, size_t ws_size,
                              hipStream_t stream)
{
  (void)in_sizes; (void)n_in; (void)out_size; (void)ws_size;
  const float* x     = (const float*)d_in[0];
  const float* y     = (const float*)d_in[1];
  const float* w_in1 = (const float*)d_in[2];
  const float* b_in1 = (const float*)d_in[3];
  const float* w_in2 = (const float*)d_in[4];
  const float* b_in2 = (const float*)d_in[5];
  const float* w_mid = (const float*)d_in[6];
  const float* b_mid = (const float*)d_in[7];
  const float* w_avg = (const float*)d_in[8];
  const float* b_avg = (const float*)d_in[9];
  const float* w_fin = (const float*)d_in[10];
  const float* b_fin = (const float*)d_in[11];
  const float* w_c1  = (const float*)d_in[12];
  const float* b_c1  = (const float*)d_in[13];
  const float* w_c2  = (const float*)d_in[14];
  const float* b_c2  = (const float*)d_in[15];
  const float* w_l   = (const float*)d_in[16];
  const float* b_l   = (const float*)d_in[17];
  float* out = (float*)d_out;

  // ---- workspace layout (ping-pong activations + prep tables), ~47 MB ----
  float*    R0  = (float*)d_ws;            // 3,875,328 floats (conv1 out / mids)
  float*    R1  = R0 + 3875328;            // 7,573,504 floats (conv2 out / y1)
  float*    HT  = R1 + 7573504;            // 392 floats (h_theta, padded)
  int*      IDX = (int*)(HT + 392);        // 5,408 ints: im2col tables
  _Float16* WSZ = (_Float16*)(IDX + 5408); // 677,888 f16: swizzled weights

  static const int idxOffs[7] = {0, 96, 672, 1824, 2976, 4128, 5280};
  static const int wszOffs[7] = {0, 6144, 79872, 227328, 374784, 522240, 669696};

  auto cdiv = [](int a, int b) { return (a + b - 1) / b; };
  auto conv = [&](const float* src, const float* wt, const float* bs, float* dst,
                  int Nb, int Cin, int Hin, int Win, int Cout, int KH, int KW,
                  int stride, int relu, int li) {
    int Hout = (Hin - KH) / stride + 1;
    int Wout = (Win - KW) / stride + 1;
    int M = Nb * Hout * Wout;
    int K = Cin * KH * KW;
    int Kpad = cdiv(K, 32) * 32;
    int nChunks = Kpad / 32;
    int nTiles = cdiv(Cout, 16);
    int gy = cdiv(nTiles, 4);
    int nTilesPad = gy * 4;
    int* idxP = IDX + idxOffs[li];
    _Float16* wzP = WSZ + wszOffs[li];
    build_idx_kernel<<<cdiv(Kpad, 128), 128, 0, stream>>>(
        idxP, K, Kpad, KH * KW, KW, Hin, Win);
    int tot = nTilesPad * nChunks * 512;
    build_wswz_kernel<<<cdiv(tot, 256), 256, 0, stream>>>(
        wzP, wt, Cout, K, nChunks, nTilesPad);
    dim3 grid(cdiv(M, 32), gy);
    conv_wmma_tiled<<<grid, 128, 0, stream>>>(
        src, idxP, wzP, bs, dst, M, Hout * Wout, Wout, Cout, nChunks,
        Cin * Hin * Win, Win, stride, relu);
  };
  auto inorm = [&](float* buf, int NC, int HW) {
    inorm_relu_kernel<<<NC, 256, 0, stream>>>(buf, HW);
  };

  // ---- coefficient branch ----
  conv(x,  w_in1, b_in1, R0, 2,   3, 352, 352,  64, 5, 5, 2, 1, 0); // (2,64,174,174)+relu
  conv(R0, w_in2, b_in2, R1, 2,  64, 174, 174, 128, 3, 3, 1, 0, 1); // (2,128,172,172)
  inorm(R1, 256, 172 * 172);
  conv(R1, w_mid + 0 * 147456, b_mid + 0,   R0, 2, 128, 172, 172, 128, 3, 3, 2, 0, 2);
  inorm(R0, 256, 85 * 85);
  conv(R0, w_mid + 1 * 147456, b_mid + 128, R1, 2, 128,  85,  85, 128, 3, 3, 2, 0, 3);
  inorm(R1, 256, 42 * 42);
  conv(R1, w_mid + 2 * 147456, b_mid + 256, R0, 2, 128,  42,  42, 128, 3, 3, 2, 0, 4);
  inorm(R0, 256, 20 * 20);
  conv(R0, w_avg, b_avg, R1, 2, 128, 20, 20, 128, 3, 3, 2, 0, 5);   // (2,128,9,9)
  inorm(R1, 256, 9 * 9);                                            // relu(relu)==relu
  conv(R1, w_fin, b_fin, R0, 2, 128, 9, 9, 3, 1, 1, 1, 1, 6);       // (2,3,9,9)+relu

  head_kernel<<<1, 512, 0, stream>>>(R0, w_c1, b_c1, w_c2, b_c2, w_l, b_l, HT);

  // ---- per-pixel polynomial stages over y (512x512) ----
  int totalPix = 2 * 512 * 512;
  y1_kernel   <<<cdiv(totalPix, 256), 256, 0, stream>>>(y, HT, R1);
  final_kernel<<<cdiv(totalPix, 256), 256, 0, stream>>>(R1, y, HT, out);
}
